// Encoder_67740224193031
// MI455X (gfx1250) — matrix operand order repaired and checked by reading, not compiled, not run on hardware
//
#include <hip/hip_runtime.h>
#include <math.h>

#define TSEQ 2048
#define VD   128      // visit-embedding dim
#define HD   256      // hidden dim
#define GD   768      // 3*HD gate dim
#define KSPLIT 8      // K-chunks for the embed GEMM
#define LN_EPS 1e-5f

typedef __attribute__((ext_vector_type(16))) __bf16 v16bf;
typedef __attribute__((ext_vector_type(8)))  float  v8f;
typedef unsigned int u32x4 __attribute__((ext_vector_type(4)));
typedef int          i32x8 __attribute__((ext_vector_type(8)));
typedef int          i32x4 __attribute__((ext_vector_type(4)));

// ---------- helpers ----------
__device__ __forceinline__ unsigned short f2bf_bits(float f) {
    unsigned u = __float_as_uint(f);
    unsigned r = (u + 0x7fffu + ((u >> 16) & 1u)) >> 16;   // RNE
    return (unsigned short)r;
}
__device__ __forceinline__ float sigm(float x) { return 1.0f / (1.0f + __expf(-x)); }

// TDM: async-load one contiguous row of `nfloats` fp32 from global -> LDS.
// D# fields per CDNA5 ISA ch.8 (group0: count/lds_addr/global_addr/type,
// group1: data_size=4B, tensor_dim0=tile_dim0=nfloats, 1 row).
// This toolchain exposes the 6-arg builtin (g0, g1, g2, g3, extra, cpol).
__device__ __forceinline__ void tdm_load_row(const float* gsrc, unsigned lds_off,
                                             int nfloats) {
    unsigned long long ga = (unsigned long long)(uintptr_t)gsrc;
    u32x4 g0;
    g0.x = 1u;                                            // count=1 (valid user D#)
    g0.y = lds_off;                                       // LDS byte address
    g0.z = (unsigned)(ga & 0xffffffffu);                  // global_addr[31:0]
    g0.w = (unsigned)((ga >> 32) & 0x01ffffffu) | (2u << 30); // addr[56:32], type=2
    i32x8 g1;
    g1[0] = (2 << 16);                                    // data_size=4B, mask/flags=0
    g1[1] = (nfloats & 0xffff) << 16;                     // tensor_dim0[15:0] @bit48
    g1[2] = ((nfloats >> 16) & 0xffff) | (1 << 16);       // dim0 hi, tensor_dim1=1
    g1[3] = (nfloats & 0xffff) << 16;                     // tile_dim0 @bit112
    g1[4] = 0;                                            // tile_dim1/2 unused
    g1[5] = nfloats;                                      // tensor_dim0_stride lo32
    g1[6] = 0;
    g1[7] = 0;
    i32x4 z4 = {0, 0, 0, 0};                              // groups 2/3 (unused, <=2D)
    i32x8 z8 = {0, 0, 0, 0, 0, 0, 0, 0};
    __builtin_amdgcn_tensor_load_to_lds(g0, g1, z4, z4, z8, 0);
}

// =====================================================================
// Kernel 1: partial ve = H^T @ X over one K-chunk (bf16 WMMA, fp32 out)
// grid (16, KSPLIT, 3), block 256 (8 waves; wave = 16 rows x full N=128)
// =====================================================================
__global__ __launch_bounds__(256)
void embed_gemm_kernel(const float* __restrict__ H0, const float* __restrict__ H1,
                       const float* __restrict__ H2,
                       const float* __restrict__ X0, const float* __restrict__ X1,
                       const float* __restrict__ X2,
                       int V0, int V1, int V2,
                       float* __restrict__ part)   // [3][KSPLIT][TSEQ][VD]
{
    const int br = blockIdx.z;
    const int kc = blockIdx.y;
    const float* __restrict__ H = (br == 0) ? H0 : (br == 1) ? H1 : H2;
    const float* __restrict__ X = (br == 0) ? X0 : (br == 1) ? X1 : X2;
    const int V = (br == 0) ? V0 : (br == 1) ? V1 : V2;

    const int chunk = ((V / KSPLIT) + 31) & ~31;
    const int kbeg  = kc * chunk;
    const int kend  = (kbeg + chunk < V) ? (kbeg + chunk) : V;

    const int w    = threadIdx.x >> 5;
    const int lane = threadIdx.x & 31;
    const int lm   = lane & 15;
    const int lhi  = lane >> 4;
    const int m0   = blockIdx.x * 128 + w * 16;

    v8f acc[8];
    #pragma unroll
    for (int i = 0; i < 8; ++i)
        #pragma unroll
        for (int j = 0; j < 8; ++j) acc[i][j] = 0.0f;

    const int kfull = kbeg + ((kend - kbeg) & ~31);
    for (int k0 = kbeg; k0 < kfull; k0 += 32) {
        // ---- A fragment: 16x32 bf16, contiguous k = 16*lhi + e per lane half ----
        const float* aBase = H + (size_t)(k0 + 16 * lhi) * TSEQ + (m0 + lm);
        __builtin_prefetch(aBase + (size_t)32 * TSEQ, 0, 1);  // next K tile
        v16bf a;
        #pragma unroll
        for (int e = 0; e < 16; ++e) {
            const int krow = e;
            a[e] = (__bf16)aBase[(size_t)krow * TSEQ];
        }
        // ---- B fragments: 32x16 bf16 each, 8 tiles = full N=128 ----
        const float* bBase = X + (size_t)(k0 + 16 * lhi) * VD + lm;
        #pragma unroll
        for (int nt = 0; nt < 8; ++nt) {
            v16bf b;
            #pragma unroll
            for (int e = 0; e < 16; ++e)
                b[e] = (__bf16)bBase[e * VD + nt * 16];
            acc[nt] = __builtin_amdgcn_wmma_f32_16x16x32_bf16(
                false, a, false, b, (short)0, acc[nt], false, false);
        }
    }
    if (kfull < kend) {                      // K tail (proc branch last chunk)
        const int k0 = kfull;
        v16bf a;
        #pragma unroll
        for (int e = 0; e < 16; ++e) {
            const int kk = k0 + 16 * lhi + e;
            a[e] = (kk < kend) ? (__bf16)H[(size_t)kk * TSEQ + m0 + lm] : (__bf16)0.0f;
        }
        #pragma unroll
        for (int nt = 0; nt < 8; ++nt) {
            v16bf b;
            #pragma unroll
            for (int e = 0; e < 16; ++e) {
                const int kk = k0 + 16 * lhi + e;
                b[e] = (kk < kend) ? (__bf16)X[(size_t)kk * VD + nt * 16 + lm]
                                   : (__bf16)0.0f;
            }
            acc[nt] = __builtin_amdgcn_wmma_f32_16x16x32_bf16(
                false, a, false, b, (short)0, acc[nt], false, false);
        }
    }
    // ---- store fp32 partial; C layout: VGPR g => row m0+g+8*lhi, col lane%16 ----
    float* __restrict__ po = part + ((size_t)br * KSPLIT + kc) * (TSEQ * VD);
    #pragma unroll
    for (int nt = 0; nt < 8; ++nt)
        #pragma unroll
        for (int g = 0; g < 8; ++g) {
            const int row = m0 + g + 8 * lhi;
            const int col = nt * 16 + lm;
            po[(size_t)row * VD + col] = acc[nt][g];
        }
}

// =====================================================================
// Kernel 1b: deterministic reduce of K-split partials -> ve bf16
// =====================================================================
__global__ __launch_bounds__(256)
void reduce_ve_kernel(const float* __restrict__ part,
                      unsigned short* __restrict__ ve)
{
    const size_t idx = (size_t)blockIdx.x * 256 + threadIdx.x;  // < 3*TSEQ*VD
    const size_t br  = idx / (TSEQ * VD);
    const size_t rem = idx % (TSEQ * VD);
    float s = 0.0f;
    #pragma unroll
    for (int c = 0; c < KSPLIT; ++c)
        s += part[(br * KSPLIT + c) * (size_t)(TSEQ * VD) + rem];
    ve[idx] = f2bf_bits(s);
}

// =====================================================================
// Kernel 2: xp = ve @ Wih^T + bih (bf16 WMMA), fp32 [3][TSEQ][GD]
// grid (64, 3), block 256: waves = 2 (M) x 4 (N groups of 192)
// =====================================================================
__global__ __launch_bounds__(256)
void proj_gemm_kernel(const unsigned short* __restrict__ ve,
                      const float* __restrict__ Wih0, const float* __restrict__ Wih1,
                      const float* __restrict__ Wih2,
                      const float* __restrict__ bih0, const float* __restrict__ bih1,
                      const float* __restrict__ bih2,
                      float* __restrict__ xp)
{
    const int br = blockIdx.y;
    const float* __restrict__ Wih = (br == 0) ? Wih0 : (br == 1) ? Wih1 : Wih2;
    const float* __restrict__ bih = (br == 0) ? bih0 : (br == 1) ? bih1 : bih2;
    const unsigned short* __restrict__ veb = ve + (size_t)br * TSEQ * VD;
    float* __restrict__ xpo = xp + (size_t)br * TSEQ * GD;

    const int w    = threadIdx.x >> 5;
    const int lane = threadIdx.x & 31;
    const int lm   = lane & 15;
    const int lhi  = lane >> 4;
    const int wm   = w & 1;
    const int wn   = w >> 1;
    const int m0   = blockIdx.x * 32 + wm * 16;
    const int n0   = wn * 192;

    v8f acc[12];
    #pragma unroll
    for (int i = 0; i < 12; ++i)
        #pragma unroll
        for (int j = 0; j < 8; ++j) acc[i][j] = 0.0f;

    #pragma unroll
    for (int k0 = 0; k0 < VD; k0 += 32) {
        const unsigned short* ap = veb + (size_t)(m0 + lm) * VD + k0 + 16 * lhi;
        union { uint4 q[2]; v16bf v; } af;
        af.q[0] = *(const uint4*)(ap);
        af.q[1] = *(const uint4*)(ap + 8);
        #pragma unroll
        for (int nt = 0; nt < 12; ++nt) {
            const int n = n0 + nt * 16 + lm;
            const float* bp = Wih + (size_t)n * VD + k0 + 16 * lhi;
            const float4 f0 = ((const float4*)bp)[0];
            const float4 f1 = ((const float4*)bp)[1];
            const float4 f2 = ((const float4*)bp)[2];
            const float4 f3 = ((const float4*)bp)[3];
            v16bf b;
            b[0]  = (__bf16)f0.x; b[1]  = (__bf16)f0.y; b[2]  = (__bf16)f0.z; b[3]  = (__bf16)f0.w;
            b[4]  = (__bf16)f1.x; b[5]  = (__bf16)f1.y; b[6]  = (__bf16)f1.z; b[7]  = (__bf16)f1.w;
            b[8]  = (__bf16)f2.x; b[9]  = (__bf16)f2.y; b[10] = (__bf16)f2.z; b[11] = (__bf16)f2.w;
            b[12] = (__bf16)f3.x; b[13] = (__bf16)f3.y; b[14] = (__bf16)f3.z; b[15] = (__bf16)f3.w;
            acc[nt] = __builtin_amdgcn_wmma_f32_16x16x32_bf16(
                false, af.v, false, b, (short)0, acc[nt], false, false);
        }
    }
    #pragma unroll
    for (int nt = 0; nt < 12; ++nt) {
        const int col  = n0 + nt * 16 + lm;
        const float bb = bih[col];
        #pragma unroll
        for (int g = 0; g < 8; ++g) {
            const int row = m0 + g + 8 * lhi;
            xpo[(size_t)row * GD + col] = acc[nt][g] + bb;
        }
    }
}

// =====================================================================
// Kernel 3: persistent GRU, one block per branch, 768 threads.
// Whh register-resident (bf16 pairs); next xp row double-buffered into
// LDS by the Tensor Data Mover (wave 0 issues, s_wait_tensorcnt syncs).
// =====================================================================
__global__ __launch_bounds__(768, 1)
void gru_kernel(const float* __restrict__ xp,
                const float* __restrict__ Whh0, const float* __restrict__ Whh1,
                const float* __restrict__ Whh2,
                const float* __restrict__ bhh0, const float* __restrict__ bhh1,
                const float* __restrict__ bhh2,
                float* __restrict__ hs)
{
    const int br = blockIdx.x;
    const float* __restrict__ Whh = (br == 0) ? Whh0 : (br == 1) ? Whh1 : Whh2;
    const float* __restrict__ bhh = (br == 0) ? bhh0 : (br == 1) ? bhh1 : bhh2;
    const float* __restrict__ xpb = xp + (size_t)br * TSEQ * GD;
    float* __restrict__ hsb = hs + (size_t)br * TSEQ * HD;

    const int o = threadIdx.x;            // gate-row 0..767

    // pack this row of Whh into 128 dwords of bf16 pairs (register resident)
    unsigned wpk[HD / 2];
    {
        const float2* wr = (const float2*)(Whh + (size_t)o * HD);
        #pragma unroll
        for (int j = 0; j < HD / 2; ++j) {
            const float2 ww = wr[j];
            wpk[j] = (unsigned)f2bf_bits(ww.x) | ((unsigned)f2bf_bits(ww.y) << 16);
        }
    }

    __shared__ float hsh[HD];
    __shared__ float gh[GD];
    __shared__ float xtb[2][GD];          // TDM double buffer for xp rows
    if (o < HD) hsh[o] = 0.0f;
    const float bias = bhh[o];

    const unsigned lds0 = (unsigned)(uintptr_t)&xtb[0][0];
    const unsigned lds1 = (unsigned)(uintptr_t)&xtb[1][0];
    if (o < 32) tdm_load_row(xpb, lds0, GD);      // t=0 row
    __syncthreads();

    #pragma unroll 1
    for (int t = 0; t < TSEQ; ++t) {
        // gh[o] = Whh[o,:] . h + bhh[o]
        float a = bias;
        #pragma unroll
        for (int j = 0; j < HD / 4; ++j) {
            const float4 hv = *(const float4*)(&hsh[j * 4]);
            const unsigned p0 = wpk[2 * j], p1 = wpk[2 * j + 1];
            a = fmaf(__uint_as_float(p0 << 16),          hv.x, a);
            a = fmaf(__uint_as_float(p0 & 0xffff0000u),  hv.y, a);
            a = fmaf(__uint_as_float(p1 << 16),          hv.z, a);
            a = fmaf(__uint_as_float(p1 & 0xffff0000u),  hv.w, a);
        }
        gh[o] = a;
        if (o < 32) __builtin_amdgcn_s_wait_tensorcnt(0);   // xt[t] landed
        __syncthreads();
        if (o < 32 && (t + 1) < TSEQ)                       // prefetch xt[t+1]
            tdm_load_row(xpb + (size_t)(t + 1) * GD,
                         (((t + 1) & 1) ? lds1 : lds0), GD);
        if (o < HD) {
            const float* xt = &xtb[t & 1][0];
            const float r = sigm(xt[o]           + gh[o]);
            const float z = sigm(xt[HD + o]      + gh[HD + o]);
            const float n = tanhf(xt[2 * HD + o] + r * gh[2 * HD + o]);
            const float hnew = (1.0f - z) * n + z * hsh[o];
            hsh[o] = hnew;
            hsb[(size_t)t * HD + o] = hnew;
        }
        __syncthreads();
    }
}

// =====================================================================
// Kernel 4a: attention scores. grid (16,3), block 128; one t per thread.
// =====================================================================
__global__ __launch_bounds__(128)
void attn_score_kernel(const float* __restrict__ hs,
                       const float* __restrict__ wa0, const float* __restrict__ wa1,
                       const float* __restrict__ wa2,
                       float* __restrict__ sco)     // [3][TSEQ]
{
    const int br = blockIdx.y;
    const float* __restrict__ wa = (br == 0) ? wa0 : (br == 1) ? wa1 : wa2;
    const float* __restrict__ hb = hs + (size_t)br * TSEQ * HD;
    const int t = blockIdx.x * 128 + threadIdx.x;

    const float4* row = (const float4*)(hb + (size_t)t * HD);
    const float4* w4  = (const float4*)wa;
    float s = 0.0f;
    #pragma unroll 8
    for (int j = 0; j < HD / 4; ++j) {
        const float4 v = row[j], wv = w4[j];
        s = fmaf(v.x, wv.x, s); s = fmaf(v.y, wv.y, s);
        s = fmaf(v.z, wv.z, s); s = fmaf(v.w, wv.w, s);
    }
    sco[(size_t)br * TSEQ + t] = s;
}

// =====================================================================
// Kernel 4b: softmax (redundant per block, scores are tiny) + partial
// weighted pool over a 128-step slice. grid (16,3), block 256.
// =====================================================================
__global__ __launch_bounds__(256)
void attn_part_kernel(const float* __restrict__ hs,
                      const float* __restrict__ sco,
                      float* __restrict__ ppool)    // [3][16][HD]
{
    const int br  = blockIdx.y;
    const int blk = blockIdx.x;
    const float* __restrict__ hb = hs + (size_t)br * TSEQ * HD;
    const int tid = threadIdx.x;

    __shared__ float ssc[TSEQ];
    __shared__ float red[256];

    for (int i = tid; i < TSEQ; i += 256) ssc[i] = sco[(size_t)br * TSEQ + i];
    __syncthreads();

    float mx = -3.4e38f;
    for (int i = tid; i < TSEQ; i += 256) mx = fmaxf(mx, ssc[i]);
    red[tid] = mx;
    __syncthreads();
    for (int s = 128; s > 0; s >>= 1) {
        if (tid < s) red[tid] = fmaxf(red[tid], red[tid + s]);
        __syncthreads();
    }
    mx = red[0];
    __syncthreads();

    float lsum = 0.0f;
    for (int i = tid; i < TSEQ; i += 256) {
        const float e = __expf(ssc[i] - mx);
        ssc[i] = e;
        lsum += e;
    }
    red[tid] = lsum;
    __syncthreads();
    for (int s = 128; s > 0; s >>= 1) {
        if (tid < s) red[tid] += red[tid + s];
        __syncthreads();
    }
    const float tot = red[0];
    __syncthreads();

    // partial pool over this block's 128-step slice
    const int t0 = blk * 128;
    float acc = 0.0f;
    #pragma unroll 4
    for (int tt = 0; tt < 128; ++tt)
        acc = fmaf(ssc[t0 + tt], hb[(size_t)(t0 + tt) * HD + tid], acc);
    ppool[((size_t)br * 16 + blk) * HD + tid] = acc / tot;
}

// =====================================================================
// Kernel 5: reduce partial pools + MLP head + LayerNorm. 1 block, 768 thr.
// =====================================================================
__global__ __launch_bounds__(768)
void head_kernel(const float* __restrict__ ppool,
                 const float* __restrict__ W1, const float* __restrict__ b1,
                 const float* __restrict__ W2, const float* __restrict__ b2,
                 const float* __restrict__ ln_g, const float* __restrict__ ln_b,
                 float* __restrict__ out)
{
    __shared__ float hc[GD];
    __shared__ float h1s[2 * HD];
    __shared__ float h2s[HD];
    __shared__ float red[256];
    const int tid = threadIdx.x;

    {   // h_cat[tid] = sum over 16 block-partials
        const int br = tid / HD, j = tid % HD;
        float s = 0.0f;
        #pragma unroll
        for (int b = 0; b < 16; ++b)
            s += ppool[((size_t)br * 16 + b) * HD + j];
        hc[tid] = s;
    }
    __syncthreads();

    if (tid < 2 * HD) {
        const float4* wr = (const float4*)(W1 + (size_t)tid * GD);
        const float4* hv = (const float4*)hc;
        float s = b1[tid];
        #pragma unroll 8
        for (int j = 0; j < GD / 4; ++j) {
            const float4 a = wr[j], b = hv[j];
            s = fmaf(a.x, b.x, s); s = fmaf(a.y, b.y, s);
            s = fmaf(a.z, b.z, s); s = fmaf(a.w, b.w, s);
        }
        h1s[tid] = fmaxf(s, 0.0f);
    }
    __syncthreads();
    if (tid < HD) {
        const float4* wr = (const float4*)(W2 + (size_t)tid * (2 * HD));
        const float4* hv = (const float4*)h1s;
        float s = b2[tid];
        #pragma unroll 8
        for (int j = 0; j < (2 * HD) / 4; ++j) {
            const float4 a = wr[j], b = hv[j];
            s = fmaf(a.x, b.x, s); s = fmaf(a.y, b.y, s);
            s = fmaf(a.z, b.z, s); s = fmaf(a.w, b.w, s);
        }
        h2s[tid] = s;
    }
    __syncthreads();

    if (tid < 256) red[tid] = h2s[tid];
    __syncthreads();
    for (int s = 128; s > 0; s >>= 1) {
        if (tid < s) red[tid] += red[tid + s];
        __syncthreads();
    }
    const float mu = red[0] * (1.0f / HD);
    __syncthreads();
    if (tid < 256) { const float d = h2s[tid] - mu; red[tid] = d * d; }
    __syncthreads();
    for (int s = 128; s > 0; s >>= 1) {
        if (tid < s) red[tid] += red[tid + s];
        __syncthreads();
    }
    const float var = red[0] * (1.0f / HD);

    if (tid < HD)
        out[tid] = (h2s[tid] - mu) * rsqrtf(var + LN_EPS) * ln_g[tid] + ln_b[tid];
}

// =====================================================================
extern "C" void kernel_launch(void* const* d_in, const int* in_sizes, int n_in,
                              void* d_out, int out_size, void* d_ws, size_t ws_size,
                              hipStream_t stream) {
    const float* H_diag = (const float*)d_in[0];
    const float* H_med  = (const float*)d_in[1];
    const float* H_proc = (const float*)d_in[2];
    const float* Xd     = (const float*)d_in[3];
    const float* Xm     = (const float*)d_in[4];
    const float* Xp     = (const float*)d_in[5];
    const float* Wih_d  = (const float*)d_in[6];
    const float* Whh_d  = (const float*)d_in[7];
    const float* bih_d  = (const float*)d_in[8];
    const float* bhh_d  = (const float*)d_in[9];
    const float* Wih_m  = (const float*)d_in[10];
    const float* Whh_m  = (const float*)d_in[11];
    const float* bih_m  = (const float*)d_in[12];
    const float* bhh_m  = (const float*)d_in[13];
    const float* Wih_p  = (const float*)d_in[14];
    const float* Whh_p  = (const float*)d_in[15];
    const float* bih_p  = (const float*)d_in[16];
    const float* bhh_p  = (const float*)d_in[17];
    const float* wa_d   = (const float*)d_in[18];
    const float* wa_m   = (const float*)d_in[19];
    const float* wa_p   = (const float*)d_in[20];
    const float* W1     = (const float*)d_in[21];
    const float* b1     = (const float*)d_in[22];
    const float* W2     = (const float*)d_in[23];
    const float* b2     = (const float*)d_in[24];
    const float* ln_g   = (const float*)d_in[25];
    const float* ln_b   = (const float*)d_in[26];

    // workspace layout (parts region later reused for xp)
    char* ws = (char*)d_ws;
    size_t off = 0;
    unsigned short* ve = (unsigned short*)(ws + off);
    off += (size_t)3 * TSEQ * VD * sizeof(unsigned short);
    off = (off + 255) & ~(size_t)255;
    float* part = (float*)(ws + off);                 // [3][KSPLIT][TSEQ][VD]
    float* xp   = part;                               // alias: xp written after reduce
    off += (size_t)3 * KSPLIT * TSEQ * VD * sizeof(float);   // >= 3*TSEQ*GD*4
    off = (off + 255) & ~(size_t)255;
    float* hs = (float*)(ws + off);
    off += (size_t)3 * TSEQ * HD * sizeof(float);
    off = (off + 255) & ~(size_t)255;
    float* sco = (float*)(ws + off);
    off += (size_t)3 * TSEQ * sizeof(float);
    off = (off + 255) & ~(size_t)255;
    float* ppool = (float*)(ws + off);
    off += (size_t)3 * 16 * HD * sizeof(float);
    (void)ws_size; (void)in_sizes; (void)n_in; (void)out_size;

    dim3 gE(16, KSPLIT, 3);
    embed_gemm_kernel<<<gE, 256, 0, stream>>>(H_diag, H_med, H_proc, Xd, Xm, Xp,
                                              8000, 4000, 2000, part);
    reduce_ve_kernel<<<(3 * TSEQ * VD) / 256, 256, 0, stream>>>(part, ve);
    dim3 gP(64, 3);
    proj_gemm_kernel<<<gP, 256, 0, stream>>>(ve, Wih_d, Wih_m, Wih_p,
                                             bih_d, bih_m, bih_p, xp);
    gru_kernel<<<3, 768, 0, stream>>>(xp, Whh_d, Whh_m, Whh_p,
                                      bhh_d, bhh_m, bhh_p, hs);
    dim3 gS(16, 3);
    attn_score_kernel<<<gS, 128, 0, stream>>>(hs, wa_d, wa_m, wa_p, sco);
    attn_part_kernel<<<gS, 256, 0, stream>>>(hs, sco, ppool);
    head_kernel<<<1, 768, 0, stream>>>(ppool, W1, b1, W2, b2, ln_g, ln_b,
                                       (float*)d_out);
}
